// Groups_18915035972343
// MI455X (gfx1250) — matrix operand order repaired and checked by reading, not compiled, not run on hardware
//
#include <hip/hip_runtime.h>
#include <hip/hip_bf16.h>

#define RUNS 512
#define WAYS 5
#define SHOT 5
#define QUERIES 15
#define DIM 640
#define NS (WAYS*SHOT)       // 25
#define QS (WAYS*QUERIES)    // 75
#define NTOT (NS+QS)         // 100
#define NAUG (NTOT+WAYS)     // 105
#define ASTRIDE 106          // LDS row stride for the 105x105 matrix
#define LAM 10.0f
#define ALPHA_ 0.7f
#define UPDATE_RATE 0.6f
#define EPOCHS 3
#define OT_EPS_ 0.001f
#define OT_MAXITERS 1000
#define LOG5 1.6094379124341003f

#define NPADF 112            // padded (zero) feature rows for WMMA tiling
#define KBLK 32              // K per WMMA step
#define NKB (DIM/KBLK)       // 20
#define PANEL_B (NPADF*KBLK*2)   // 7168 bytes per K-panel (f16)

// per-run workspace byte offsets
#define OFF_F32   0
#define OFF_F16   (NPADF*DIM*4)             // 286720
#define OFF_PROTO (OFF_F16 + NPADF*DIM*2)   // 430080
#define OFF_N2    (OFF_PROTO + WAYS*DIM*4)  // 442880
#define RUN_STRIDE 443392

typedef _Float16 h16v  __attribute__((ext_vector_type(16)));
typedef float    f32x8 __attribute__((ext_vector_type(8)));
typedef int      v4i_t __attribute__((ext_vector_type(4)));

// ---- CDNA5 async copy to LDS (ASYNCcnt-tracked DMA, no VGPR round-trip) ----
#if defined(__HIP_DEVICE_COMPILE__) && __has_builtin(__builtin_amdgcn_global_load_async_to_lds_b128)
typedef __attribute__((address_space(1))) v4i_t gv4i;
typedef __attribute__((address_space(3))) v4i_t lv4i;
#define ASYNC_CP16(g, l) \
  __builtin_amdgcn_global_load_async_to_lds_b128((gv4i*)(void*)(g), (lv4i*)(void*)(l), 0, 0)
#else
#define ASYNC_CP16(g, l) \
  asm volatile("global_load_async_to_lds_b128 %0, %1, off" \
               :: "v"((unsigned)(unsigned long long)(l)), \
                  "v"((unsigned long long)(g)) : "memory")
#endif

#if defined(__HIP_DEVICE_COMPILE__) && __has_builtin(__builtin_amdgcn_s_wait_asynccnt)
#define ASYNC_WAIT0() __builtin_amdgcn_s_wait_asynccnt(0)
#else
#define ASYNC_WAIT0() asm volatile("s_wait_asynccnt 0x0" ::: "memory")
#endif

__device__ __forceinline__ float blk_max(float* red, int tid, float v){
  red[tid] = v; __syncthreads();
  for (int s = 128; s > 0; s >>= 1){
    if (tid < s) red[tid] = fmaxf(red[tid], red[tid+s]);
    __syncthreads();
  }
  float r = red[0]; __syncthreads(); return r;
}

__device__ __forceinline__ float blk_sum(float* red, int tid, float v){
  red[tid] = v; __syncthreads();
  for (int s = 128; s > 0; s >>= 1){
    if (tid < s) red[tid] += red[tid+s];
    __syncthreads();
  }
  float r = red[0]; __syncthreads(); return r;
}

// Sinkhorn on Z (row stride 8, WAYS cols), r = ones, c = cval, clamp first n_l
// rows to one-hot(ysr). Matches reference while_loop semantics.
__device__ void sinkhorn_lds(float* Z, int n, int n_l, float cval,
                             const int* ysr, float* red, float* cs, int tid){
  float u = 0.f;
  __syncthreads();
  for (int it = 1; ; ++it){
    float s = 0.f;
    if (tid < n){
      #pragma unroll
      for (int w = 0; w < WAYS; ++w) s += Z[tid*8+w];
    }
    float diff = (tid < n) ? fabsf(u - s) : 0.f;
    float dmax = blk_max(red, tid, diff);
    if (!((dmax > OT_EPS_) && (it <= OT_MAXITERS))) break;
    u = s;
    if (tid < n){
      float inv = 1.f / s;
      #pragma unroll
      for (int w = 0; w < WAYS; ++w) Z[tid*8+w] *= inv;
    }
    __syncthreads();
    if (tid < WAYS){
      float c_ = 0.f;
      for (int i = 0; i < n; ++i) c_ += Z[i*8+tid];
      cs[tid] = cval / c_;
    }
    __syncthreads();
    if (tid < n){
      #pragma unroll
      for (int w = 0; w < WAYS; ++w) Z[tid*8+w] *= cs[w];
      if (tid < n_l){
        int lab = ysr[tid];
        #pragma unroll
        for (int w = 0; w < WAYS; ++w) Z[tid*8+w] = (w == lab) ? 1.f : 0.f;
      }
    }
    __syncthreads();
  }
}

__global__ __launch_bounds__(256)
void fewshot_init(const float* __restrict__ xs, const float* __restrict__ xq,
                  char* __restrict__ ws){
  int run = blockIdx.x, tid = threadIdx.x;
  char* base = ws + (size_t)run * RUN_STRIDE;
  float*    F32   = (float*)(base + OFF_F32);
  _Float16* F16   = (_Float16*)(base + OFF_F16);
  float*    proto = (float*)(base + OFF_PROTO);
  float*    n2    = (float*)(base + OFF_N2);
  const float* xsr = xs + (size_t)run * NS * DIM;
  const float* xqr = xq + (size_t)run * QS * DIM;

  for (int idx = tid; idx < NPADF*DIM; idx += 256){
    int r = idx / DIM, d = idx - r*DIM;
    float v = 0.f;
    if (r < NS)        v = xsr[r*DIM + d];
    else if (r < NTOT) v = xqr[(r-NS)*DIM + d];
    F32[idx] = v;
    F16[idx] = (_Float16)v;
  }
  for (int idx = tid; idx < WAYS*DIM; idx += 256){
    int k = idx / DIM, d = idx - k*DIM;
    float s = 0.f;
    for (int sh = 0; sh < SHOT; ++sh) s += xsr[(k*SHOT+sh)*DIM + d];
    proto[idx] = s * (1.0f/SHOT);
  }
  for (int i = tid; i < NPADF; i += 256){
    float s = 0.f;
    if (i < NTOT){
      const float* fr = (i < NS) ? (xsr + i*DIM) : (xqr + (i-NS)*DIM);
      for (int d = 0; d < DIM; ++d){ float v = fr[d]; s += v*v; }
    }
    n2[i] = s;
  }
}

__global__ __launch_bounds__(256)
void fewshot_main(const int* __restrict__ ys, const int* __restrict__ yq,
                  char* __restrict__ ws, float* __restrict__ out){
  __shared__ float Amat[NAUG*ASTRIDE];        // 44520 B: Pq / Gram / dist / A / Inv
  __shared__ char  panelc[2*PANEL_B];         // 14336 B: double-buffered f16 K-panel
  __shared__ float Zl[NPADF*8];               //  3584 B
  __shared__ float red[256];                  //  1024 B
  __shared__ float tmp112[NPADF];             //   448 B: ent (pre-Gram) / dm (post-Gram)
  __shared__ float dps[32];
  __shared__ float cs[8];
  __shared__ float pn2s[8];
  __shared__ float msk[8];
  __shared__ float scal[8];

  int run = blockIdx.x, tid = threadIdx.x;
  int lane = tid & 31, wid = tid >> 5;
  char* base = ws + (size_t)run * RUN_STRIDE;
  float*    F32   = (float*)(base + OFF_F32);
  _Float16* F16   = (_Float16*)(base + OFF_F16);
  float*    proto = (float*)(base + OFF_PROTO);
  float*    n2    = (float*)(base + OFF_N2);
  const int* ysr = ys + run * NS;
  const int* yqr = yq + run * QS;

  // stage one 112x32 f16 K-panel into LDS buffer `buf` via async DMA
  auto stage_panel = [&](int blk, int buf){
    #pragma unroll
    for (int rep = 0; rep < 2; ++rep){
      int u = tid + rep*256;                   // 448 x 16B chunks
      if (u < (NPADF*4)){
        int r = u >> 2, part = u & 3;
        char* g = (char*)(F16 + (size_t)r*DIM + blk*KBLK) + part*16;
        char* l = panelc + buf*PANEL_B + r*(KBLK*2) + part*16;
        ASYNC_CP16(g, l);
      }
    }
  };

  for (int ep = 0; ep < EPOCHS; ++ep){
    // ---------- proto squared norms ----------
    if (tid < WAYS){
      float s = 0.f; const float* pr = proto + tid*DIM;
      for (int d = 0; d < DIM; ++d) s += pr[d]*pr[d];
      pn2s[tid] = s;
    }
    __syncthreads();

    // ---------- get_prob: Pq staged inside Amat (dead at epoch start) ----------
    float* Pq = Amat;
    for (int p = tid; p < QS*WAYS; p += 256){
      int q = p / WAYS, w = p - q*WAYS;
      const float* fr = F32 + (NS+q)*DIM;
      const float* pr = proto + w*DIM;
      float dot = 0.f;
      for (int d = 0; d < DIM; ++d) dot += fr[d]*pr[d];
      float d2 = fmaxf(n2[NS+q] + pn2s[w] - 2.f*dot, 0.f);
      Pq[q*8+w] = __expf(-LAM*d2);
    }
    __syncthreads();
    sinkhorn_lds(Pq, QS, 0, (float)(QS/WAYS), ysr, red, cs, tid);

    // assemble Z (100 rows) into Zl
    for (int i = tid; i < NTOT; i += 256){
      #pragma unroll
      for (int w = 0; w < WAYS; ++w)
        Zl[i*8+w] = (i < NS) ? ((ysr[i]==w) ? 1.f : 0.f) : Pq[(i-NS)*8+w];
    }
    __syncthreads();

    // ---------- entropy weighting (ent lives in tmp112 until omega) ----------
    if (tid < NTOT){
      float pv[WAYS]; float s = 0.f;
      #pragma unroll
      for (int w = 0; w < WAYS; ++w){ pv[w] = Zl[tid*8+w] + 1e-12f; s += pv[w]; }
      float H = 0.f;
      #pragma unroll
      for (int w = 0; w < WAYS; ++w){ float q = pv[w]/s; H -= q*__logf(q); }
      tmp112[tid] = H * (1.0f/LOG5);
    }
    __syncthreads();
    if (tid < NTOT){
      float wgt = 1.f - tmp112[tid];
      #pragma unroll
      for (int w = 0; w < WAYS; ++w) Zl[tid*8+w] *= wgt;
    }
    __syncthreads();

    // ---------- new prototypes + EMA ----------
    if (tid < WAYS){
      float s = 0.f;
      for (int i = 0; i < NTOT; ++i) s += Zl[i*8+tid];
      cs[tid] = s;
    }
    __syncthreads();
    for (int idx = tid; idx < WAYS*DIM; idx += 256){
      int k = idx / DIM, d = idx - k*DIM;
      float acc = 0.f;
      for (int i = 0; i < NTOT; ++i) acc += Zl[i*8+k] * F32[i*DIM+d];
      proto[idx] = (1.f-UPDATE_RATE)*proto[idx] + UPDATE_RATE*(acc/cs[k]);
    }
    __threadfence_block();
    __syncthreads();

    // ---------- proto-proto distances, score, mask ----------
    if (tid < WAYS){
      float s = 0.f; const float* pr = proto + tid*DIM;
      for (int d = 0; d < DIM; ++d) s += pr[d]*pr[d];
      pn2s[tid] = s;
    }
    __syncthreads();
    if (tid < WAYS*WAYS){
      int k = tid / WAYS, l = tid - k*WAYS;
      const float* pk = proto + k*DIM; const float* pl = proto + l*DIM;
      float dot = 0.f;
      for (int d = 0; d < DIM; ++d) dot += pk[d]*pl[d];
      dps[tid] = __expf(-LAM * fmaxf(pn2s[k]+pn2s[l]-2.f*dot, 0.f));
    }
    __syncthreads();
    if (tid < WAYS){
      float pv[WAYS]; float s = 0.f;
      #pragma unroll
      for (int l = 0; l < WAYS; ++l){ pv[l] = dps[tid*WAYS+l] + 1e-12f; s += pv[l]; }
      float H = 0.f;
      #pragma unroll
      for (int l = 0; l < WAYS; ++l){ float q = pv[l]/s; H -= q*__logf(q); }
      scal[tid] = H * (1.0f/LOG5);
    }
    float omega = blk_sum(red, tid, (tid < NTOT) ? tmp112[tid] : 0.f) * (1.0f/NTOT);
    if (tid < WAYS) msk[tid] = (scal[tid] < omega) ? 1.f : 0.f;
    __syncthreads();

    // ---------- append masked prototypes ----------
    for (int idx = tid; idx < WAYS*DIM; idx += 256){
      int k = idx / DIM, d = idx - k*DIM;
      float v = proto[idx] * msk[k];
      F32[(NTOT+k)*DIM + d] = v;
      F16[(NTOT+k)*DIM + d] = (_Float16)v;
    }
    if (tid < WAYS*WAYS){
      int k = tid / WAYS, l = tid - k*WAYS;
      Zl[(NTOT+k)*8 + l] = dps[tid] * msk[k];
    }
    if (tid < WAYS) n2[NTOT+tid] = pn2s[tid] * msk[tid];
    __threadfence_block();
    __syncthreads();

    // ---------- Gram = F * F^T via WMMA with async double-buffered K-panels ----
    {
      int m = lane & 15, hf = lane >> 4;
      f32x8 acc[7];
      #pragma unroll
      for (int j = 0; j < 7; ++j) acc[j] = (f32x8){0.f,0.f,0.f,0.f,0.f,0.f,0.f,0.f};

      stage_panel(0, 0);
      ASYNC_WAIT0();
      __syncthreads();

      for (int kbi = 0; kbi < NKB; ++kbi){
        if (kbi + 1 < NKB) stage_panel(kbi + 1, (kbi + 1) & 1);
        if (tid == 0 && kbi + 2 < NKB)
          __builtin_prefetch((const void*)(F16 + (kbi + 2)*KBLK), 0, 1);

        const _Float16* P = (const _Float16*)(panelc + (kbi & 1)*PANEL_B);
        #pragma unroll
        for (int j = 0; j < 7; ++j){
          int t = wid + 8*j;                    // wave-uniform
          if (t < 49){
            int tm = t / 7, tn = t - (t/7)*7;
            const _Float16* ra = P + (tm*16 + m)*KBLK;
            const _Float16* rb = P + (tn*16 + m)*KBLK;
            h16v a, b;
            #pragma unroll
            for (int v = 0; v < 8; ++v){
              int ka = (v < 4) ? (2*v + 8*hf) : (16 + 2*(v-4) + 8*hf);
              a[2*v]   = ra[ka];
              a[2*v+1] = ra[ka + 1];
              int kbb = ka;
              b[2*v]   = rb[kbb];
              b[2*v+1] = rb[kbb + 1];
            }
            acc[j] = __builtin_amdgcn_wmma_f32_16x16x32_f16(
                       false, a, false, b, (short)0, acc[j], false, false);
          }
        }
        ASYNC_WAIT0();      // next panel resident
        __syncthreads();    // all waves done reading current panel
      }

      #pragma unroll
      for (int j = 0; j < 7; ++j){
        int t = wid + 8*j;
        if (t < 49){
          int tm = t / 7, tn = t - (t/7)*7;
          #pragma unroll
          for (int r = 0; r < 8; ++r){
            int gi = tm*16 + r + 8*hf;
            int gj = tn*16 + m;
            if (gi < NAUG && gj < NAUG) Amat[gi*ASTRIDE + gj] = acc[j][r];
          }
        }
      }
    }
    __syncthreads();

    // ---------- Gram -> dist -> normalized graph -> A = I - alpha*W ----------
    for (int idx = tid; idx < NAUG*NAUG; idx += 256){
      int i = idx / NAUG, j = idx - i*NAUG;
      float g = Amat[i*ASTRIDE+j];
      float d2 = fmaxf(n2[i] + n2[j] - 2.f*g, 0.f);
      Amat[i*ASTRIDE+j] = (i == j) ? 0.f : __expf(-LAM*d2);
    }
    __syncthreads();
    if (tid < NAUG){
      float s = 0.f;
      for (int j = 0; j < NAUG; ++j) s += Amat[tid*ASTRIDE+j];
      tmp112[tid] = rsqrtf(s);                  // dm
    }
    __syncthreads();
    for (int idx = tid; idx < NAUG*NAUG; idx += 256){
      int i = idx / NAUG, j = idx - i*NAUG;
      float w = tmp112[i]*Amat[i*ASTRIDE+j]*tmp112[j];
      Amat[i*ASTRIDE+j] = ((i==j) ? 1.f : 0.f) - ALPHA_*w;
    }
    __syncthreads();

    // ---------- in-place Gauss-Jordan inverse (diag==1, no pivoting) ----------
    for (int p = 0; p < NAUG; ++p){
      if (tid == 0) scal[0] = 1.f / Amat[p*ASTRIDE+p];
      __syncthreads();
      float ipiv = scal[0];
      for (int j = tid; j < NAUG; j += 256) Amat[p*ASTRIDE+j] *= ipiv;
      __syncthreads();
      if (tid == 0) Amat[p*ASTRIDE+p] = ipiv;
      __syncthreads();
      if (tid < NAUG && tid != p){
        float f = Amat[tid*ASTRIDE+p];
        for (int j = 0; j < NAUG; ++j)
          if (j != p) Amat[tid*ASTRIDE+j] -= f * Amat[p*ASTRIDE+j];
        Amat[tid*ASTRIDE+p] = -f * ipiv;
      }
      __syncthreads();
    }

    // ---------- Z = Inv @ Z (accumulate in registers, then write back) ----------
    float zacc[WAYS] = {0.f,0.f,0.f,0.f,0.f};
    if (tid < NAUG){
      for (int j = 0; j < NAUG; ++j){
        float iv = Amat[tid*ASTRIDE+j];
        #pragma unroll
        for (int w = 0; w < WAYS; ++w) zacc[w] += iv * Zl[j*8+w];
      }
    }
    __syncthreads();
    if (tid < NAUG){
      #pragma unroll
      for (int w = 0; w < WAYS; ++w) Zl[tid*8+w] = zacc[w];
    }
    __syncthreads();

    sinkhorn_lds(Zl, NAUG, NS, (float)(NAUG/WAYS), ysr, red, cs, tid);
  }

  // ---------- scoring: accuracy over queries ----------
  float v = 0.f;
  if (tid < QS){
    int i = NS + tid;
    int best = 0; float bv = Zl[i*8+0];
    #pragma unroll
    for (int w = 1; w < WAYS; ++w){
      float z = Zl[i*8+w];
      if (z > bv){ bv = z; best = w; }
    }
    v = (best == yqr[tid]) ? 1.f : 0.f;
  }
  float s = blk_sum(red, tid, v);
  if (tid == 0) out[run] = s * (1.0f/QS);
}

extern "C" void kernel_launch(void* const* d_in, const int* in_sizes, int n_in,
                              void* d_out, int out_size, void* d_ws, size_t ws_size,
                              hipStream_t stream){
  const float* xs = (const float*)d_in[0];
  const float* xq = (const float*)d_in[1];
  const int*   ys = (const int*)d_in[2];
  const int*   yq = (const int*)d_in[3];
  float* out = (float*)d_out;
  char*  ws  = (char*)d_ws;
  fewshot_init<<<RUNS, 256, 0, stream>>>(xs, xq, ws);
  fewshot_main<<<RUNS, 256, 0, stream>>>(ys, yq, ws, out);
}
